// CosineMetric_57543971832353
// MI455X (gfx1250) — matrix operand order repaired and checked by reading, not compiled, not run on hardware
//
#include <hip/hip_runtime.h>
#include <hip/hip_bf16.h>

typedef _Float16 v16h __attribute__((ext_vector_type(16)));
typedef _Float16 v8h  __attribute__((ext_vector_type(8)));
typedef _Float16 v4h  __attribute__((ext_vector_type(4)));
typedef float    v8f  __attribute__((ext_vector_type(8)));

#define NROWS 8192
#define KDIM  512
#define EPS   1e-8f
#define SCALE 16.0f

// ---------------------------------------------------------------------------
// Kernel 1: row-normalize x1/x2 (f32) -> unit-norm f16 rows in workspace.
// One wave (32 lanes) per row; lane-strided float4 loads for coalescing.
// ---------------------------------------------------------------------------
__global__ __launch_bounds__(256) void cos_normalize_f16(
    const float* __restrict__ x1, const float* __restrict__ x2,
    _Float16* __restrict__ an, _Float16* __restrict__ bn)
{
    const int wave = (int)((blockIdx.x * blockDim.x + threadIdx.x) >> 5);
    const int lane = (int)(threadIdx.x & 31);

    const float* src;
    _Float16*    dst;
    if (wave < NROWS) {
        src = x1 + (size_t)wave * KDIM;
        dst = an + (size_t)wave * KDIM;
    } else {
        const int r = wave - NROWS;
        src = x2 + (size_t)r * KDIM;
        dst = bn + (size_t)r * KDIM;
    }

    const float4* s4 = (const float4*)src;
    float4 v[4];
    float ss = 0.0f;
#pragma unroll
    for (int j = 0; j < 4; ++j) {
        v[j] = s4[lane + 32 * j];               // 128 contiguous floats per j-step
        ss += v[j].x * v[j].x + v[j].y * v[j].y + v[j].z * v[j].z + v[j].w * v[j].w;
    }
    // wave32 butterfly reduction
#pragma unroll
    for (int off = 16; off > 0; off >>= 1)
        ss += __shfl_xor(ss, off, 32);

    const float inv = 1.0f / fmaxf(sqrtf(ss), EPS);

    v4h* d4 = (v4h*)dst;
#pragma unroll
    for (int j = 0; j < 4; ++j) {
        v4h h;
        h[0] = (_Float16)(v[j].x * inv);
        h[1] = (_Float16)(v[j].y * inv);
        h[2] = (_Float16)(v[j].z * inv);
        h[3] = (_Float16)(v[j].w * inv);
        d4[lane + 32 * j] = h;                  // 8B per lane, coalesced
    }
}

// ---------------------------------------------------------------------------
// Kernel 2: out[8192,8192] = SCALE * A(f16) @ B(f16)^T via v_wmma_f32_16x16x32_f16
// Block = 256 threads = 8 waves arranged 2(m) x 4(n); wave tile = 64x64
// (4x4 WMMA 16x16 tiles); block tile = 128x256.
//
// Fragment layouts used here (wave32; contiguous k, same order for A and B):
//   A 16x32 f16 : lane<16 row=lane,    elems = K k0..k0+15 (contiguous)
//                 lane>=16 row=lane-16, elems = K k0+16..k0+31
//   B 32x16 f16 : lane<16 col=lane,    elems = K k0..k0+15 (contiguous)
//                 lane>=16 col=lane-16, elems = K k0+16..k0+31
//   C/D 16x16 f32: VGPR r -> M=r (lane<16) / M=r+8 (lane>=16), N=lane&15
//
// All 8 fragments (4 A + 4 B) are preloaded per K-step so the scheduler can
// issue staggered s_wait_loadcnt instead of full drains between WMMA groups.
// ---------------------------------------------------------------------------
__global__ __launch_bounds__(256) void cos_gemm_wmma(
    const _Float16* __restrict__ A, const _Float16* __restrict__ B,
    float* __restrict__ out)
{
    const int lane = (int)(threadIdx.x & 31);
    const int wave = (int)(threadIdx.x >> 5);
    const int lh   = lane & 15;     // row/col within tile
    const int hs   = lane >> 4;     // half-wave select
    const int wm   = wave >> 2;     // 0..1
    const int wn   = wave & 3;      // 0..3

    const int m0 = (int)blockIdx.y * 128 + wm * 64;
    const int n0 = (int)blockIdx.x * 256 + wn * 64;

    v8f acc[4][4] = {};

    // Per-lane base pointers (element units)
    const _Float16* Abase = A + (size_t)(m0 + lh) * KDIM + hs * 16;
    const _Float16* Bbase = B + (size_t)(n0 + lh) * KDIM + hs * 16;

    for (int k0 = 0; k0 < KDIM; k0 += 32) {
        // Unconditional speculative prefetch of next K chunk
        // (global_prefetch_b8; translation failures are silently dropped).
        __builtin_prefetch(Bbase + k0 + 32, 0, 1);
        __builtin_prefetch(Abase + k0 + 32, 0, 1);

        // Preload all B fragments (reused across the 4 M-tiles)...
        v16h bfrag[4];
#pragma unroll
        for (int nt = 0; nt < 4; ++nt)
            bfrag[nt] = *(const v16h*)(Bbase + (size_t)nt * 16 * KDIM + k0);

        // ...and all A fragments, so WMMAs can start as loads retire.
        v16h afrag[4];
#pragma unroll
        for (int mt = 0; mt < 4; ++mt) {
            const _Float16* ap = Abase + (size_t)mt * 16 * KDIM + k0;
            v8h alo = *(const v8h*)(ap);        // K k0..+7   (+16 for upper half-wave)
            v8h ahi = *(const v8h*)(ap + 8);    // K k0+8..+15 (+16)
            afrag[mt] = __builtin_shufflevector(alo, ahi,
                0, 1, 2, 3, 4, 5, 6, 7, 8, 9, 10, 11, 12, 13, 14, 15);
        }

#pragma unroll
        for (int mt = 0; mt < 4; ++mt)
#pragma unroll
            for (int nt = 0; nt < 4; ++nt)
                acc[mt][nt] = __builtin_amdgcn_wmma_f32_16x16x32_f16(
                    /*neg_a=*/false, afrag[mt], /*neg_b=*/false, bfrag[nt],
                    /*c_mod=*/(short)0, acc[mt][nt],
                    /*reuse_a=*/false, /*reuse_b=*/false);
    }

    // Epilogue: C/D layout -> coalesced 64B half-wave stores
#pragma unroll
    for (int mt = 0; mt < 4; ++mt) {
#pragma unroll
        for (int r = 0; r < 8; ++r) {
            const size_t row = (size_t)(m0 + mt * 16 + r + hs * 8) * (size_t)NROWS;
#pragma unroll
            for (int nt = 0; nt < 4; ++nt)
                out[row + (size_t)(n0 + nt * 16 + lh)] = SCALE * acc[mt][nt][r];
        }
    }
}

// ---------------------------------------------------------------------------
extern "C" void kernel_launch(void* const* d_in, const int* in_sizes, int n_in,
                              void* d_out, int out_size, void* d_ws, size_t ws_size,
                              hipStream_t stream) {
    (void)in_sizes; (void)n_in; (void)out_size; (void)ws_size;

    const float* x1 = (const float*)d_in[0];
    const float* x2 = (const float*)d_in[1];
    float* out = (float*)d_out;

    _Float16* an = (_Float16*)d_ws;                       //  8 MB
    _Float16* bn = an + (size_t)NROWS * KDIM;             // +8 MB = 16 MB total

    // 16384 rows, one wave each -> 16384 waves / 8 waves-per-block = 2048 blocks
    cos_normalize_f16<<<dim3(2048), dim3(256), 0, stream>>>(x1, x2, an, bn);

    // block tile 128(m) x 256(n)
    dim3 grid(NROWS / 256, NROWS / 128);                  // (32, 64)
    cos_gemm_wmma<<<grid, dim3(256), 0, stream>>>(an, bn, out);
}
